// SkipARC_33474975105401
// MI455X (gfx1250) — matrix operand order repaired and checked by reading, not compiled, not run on hardware
//
#include <hip/hip_runtime.h>
#include <hip/hip_bf16.h>

typedef __attribute__((ext_vector_type(16))) _Float16 v16h;
typedef __attribute__((ext_vector_type(8)))  _Float16 v8h;
typedef __attribute__((ext_vector_type(8)))  float    v8f;

#define HID 128
#define INSZ 192       // 8*8*3
#define KTOT 320       // INSZ + HID
#define GATES 512      // 4*HID
#define IMG 64
#define GSZ 8

__device__ __forceinline__ float sigmoidf_(float x) {
    return 1.0f / (1.0f + __expf(-x));
}

// ---------------------------------------------------------------------------
// Init: convert [Wih|Whh] -> fused f16 weight (512x320), bias sum, zero state.
// ---------------------------------------------------------------------------
__global__ void init_state_kernel(const float* __restrict__ Wih,
                                  const float* __restrict__ Whh,
                                  const float* __restrict__ bih,
                                  const float* __restrict__ bhh,
                                  _Float16* __restrict__ Wc,
                                  float* __restrict__ bsum,
                                  _Float16* __restrict__ xh,
                                  float* __restrict__ c,
                                  float* __restrict__ h,
                                  float* __restrict__ up,
                                  float* __restrict__ cum,
                                  int B) {
    int idx = blockIdx.x * blockDim.x + threadIdx.x;
    int stride = gridDim.x * blockDim.x;
    for (int i = idx; i < GATES * KTOT; i += stride) {
        int n = i / KTOT, k = i % KTOT;
        float w = (k < INSZ) ? Wih[n * INSZ + k] : Whh[n * HID + (k - INSZ)];
        Wc[i] = (_Float16)w;
    }
    for (int i = idx; i < GATES; i += stride) bsum[i] = bih[i] + bhh[i];
    for (int i = idx; i < B * KTOT; i += stride) xh[i] = (_Float16)0.0f;
    for (int i = idx; i < B * HID; i += stride) { c[i] = 0.0f; h[i] = 0.0f; }
    for (int i = idx; i < B; i += stride) { up[i] = 1.0f; cum[i] = 0.0f; }
}

// ---------------------------------------------------------------------------
// Glimpse: per-batch-item attention read.  gparams = tanh(h @ Wg.T + bg),
// build normalized Cauchy filterbanks Fh/Fw (8x64 each) in LDS, contract
// Fh * img * Fw.T -> (3,8,8), write f16 into xh[:, 0:192].
// One 64-thread block per batch item (2 waves).
// ---------------------------------------------------------------------------
__global__ __launch_bounds__(64) void glimpse_kernel(
        const float* __restrict__ images,   // (B,2,3,64,64)
        int sel,
        const float* __restrict__ h,        // (B,128)
        const float* __restrict__ Wg,       // (3,128)
        const float* __restrict__ bg,       // (3,)
        _Float16* __restrict__ xh)          // (B,320)
{
    const int b = blockIdx.x;
    const int tid = threadIdx.x;

    __shared__ float sred[64];
    __shared__ float gpar[3];
    __shared__ float snorm[16];
    __shared__ float Fh[GSZ][IMG];
    __shared__ float Fw[GSZ][IMG];
    __shared__ float tmp[3][GSZ][IMG];

    // ---- gparams = tanh(h @ Wg.T + bg) ----
    const float* hrow = h + (size_t)b * HID;
    for (int j = 0; j < 3; ++j) {
        float p = hrow[tid] * Wg[j * HID + tid] +
                  hrow[64 + tid] * Wg[j * HID + 64 + tid];
        sred[tid] = p;
        __syncthreads();
        for (int s = 32; s > 0; s >>= 1) {
            if (tid < s) sred[tid] += sred[tid + s];
            __syncthreads();
        }
        if (tid == 0) gpar[j] = tanhf(sred[0] + bg[j]);
        __syncthreads();
    }

    // ---- filterbanks (thread tid handles image position y = tid) ----
    const float dd = gpar[2];
    const float center_h = (IMG - 1) * (gpar[0] + 1.0f) * 0.5f;
    const float center_w = (IMG - 1) * (gpar[1] + 1.0f) * 0.5f;
    const float delta = ((float)IMG / (float)GSZ) * (1.0f - fabsf(dd));
    const float inv_gamma = __expf(2.0f * fabsf(dd) - 1.0f);  // 1/gamma
    const float y = (float)tid;
    #pragma unroll
    for (int g = 0; g < GSZ; ++g) {
        float gp = delta * ((float)g - 3.5f);
        float fh = (y - (gp + center_h)) * inv_gamma;
        Fh[g][tid] = 1.0f / (1.0f + fh * fh);
        float fw = (y - (gp + center_w)) * inv_gamma;
        Fw[g][tid] = 1.0f / (1.0f + fw * fw);
    }
    __syncthreads();
    if (tid < 16) {
        int g = tid & 7;
        const float* F = (tid < 8) ? &Fh[g][0] : &Fw[g][0];
        float s = 0.0f;
        for (int yy = 0; yy < IMG; ++yy) s += F[yy];
        snorm[tid] = 1.0f / s;
    }
    __syncthreads();
    #pragma unroll
    for (int g = 0; g < GSZ; ++g) {
        Fh[g][tid] *= snorm[g];
        Fw[g][tid] *= snorm[8 + g];
    }
    __syncthreads();

    // ---- tmp[c][g][w] = sum_y Fh[g][y] * img[c][y][w]  (w = tid) ----
    const float* imgbase = images + ((size_t)b * 2 + sel) * 3 * IMG * IMG;
    for (int cch = 0; cch < 3; ++cch) {
        const float* imrow = imgbase + (size_t)cch * IMG * IMG;
        float acc[GSZ];
        #pragma unroll
        for (int g = 0; g < GSZ; ++g) acc[g] = 0.0f;
        for (int yy = 0; yy < IMG; ++yy) {
            float v = imrow[yy * IMG + tid];   // coalesced across lanes
            #pragma unroll
            for (int g = 0; g < GSZ; ++g) acc[g] += Fh[g][yy] * v;
        }
        #pragma unroll
        for (int g = 0; g < GSZ; ++g) tmp[cch][g][tid] = acc[g];
    }
    __syncthreads();

    // ---- x[c][g][k] = sum_w tmp[c][g][w] * Fw[k][w], flat idx c*64+g*8+k ----
    for (int idx = tid; idx < INSZ; idx += 64) {
        int cch = idx >> 6;
        int r = idx & 63;
        int g = r >> 3;
        int k = r & 7;
        float s = 0.0f;
        for (int w = 0; w < IMG; ++w) s += tmp[cch][g][w] * Fw[k][w];
        xh[(size_t)b * KTOT + idx] = (_Float16)s;
    }
}

// ---------------------------------------------------------------------------
// Gate GEMM:  gates(B,512) = xh(B,320)_f16 @ Wc(512,320)_f16^T + bsum
// One wave per two 16x16 output tiles (A-fragment reuse), K loop of 10 x
// v_wmma_f32_16x16x32_f16.  Block = 4 waves; grid = (B/16, 512/128).
// ---------------------------------------------------------------------------
__global__ __launch_bounds__(128) void gates_gemm_kernel(
        const _Float16* __restrict__ xh,
        const _Float16* __restrict__ Wc,
        const float* __restrict__ bsum,
        float* __restrict__ gates)
{
    const int lane = threadIdx.x & 31;
    const int wave = threadIdx.x >> 5;
    const int m0 = blockIdx.x * 16;
    const int nbase = (blockIdx.y * 4 + wave) * 32;   // two N tiles: nbase, nbase+16
    const int lr = lane & 15;
    const int hi = lane >> 4;          // 0: lanes 0-15, 1: lanes 16-31

    // A fragment as loaded here (contiguous k, same as B): lane<16 holds M=lane, K in {0..15};
    // lane>=16 holds M=lane-16, K in {16..31}.
    const _Float16* aptr  = xh + (size_t)(m0 + lr) * KTOT + hi * 16;
    // B fragment (K x N): lane holds column n = lr, K rows hi*16 .. hi*16+15.
    const _Float16* b0ptr = Wc + (size_t)(nbase + lr) * KTOT + hi * 16;
    const _Float16* b1ptr = Wc + (size_t)(nbase + 16 + lr) * KTOT + hi * 16;

    v8f acc0 = {};
    v8f acc1 = {};

    #pragma unroll
    for (int kt = 0; kt < KTOT; kt += 32) {
        if (kt + 64 < KTOT) {
            __builtin_prefetch(b0ptr + kt + 64, 0, 1);
            __builtin_prefetch(b1ptr + kt + 64, 0, 1);
        }
        v8h alo = *(const v8h*)(aptr + kt);
        v8h ahi = *(const v8h*)(aptr + kt + 8);
        v16h a;
        #pragma unroll
        for (int i = 0; i < 8; ++i) { a[i] = alo[i]; a[i + 8] = ahi[i]; }
        v16h b0v = *(const v16h*)(b0ptr + kt);
        v16h b1v = *(const v16h*)(b1ptr + kt);
        acc0 = __builtin_amdgcn_wmma_f32_16x16x32_f16(
            false, a, false, b0v, (short)0, acc0, false, false);
        acc1 = __builtin_amdgcn_wmma_f32_16x16x32_f16(
            false, a, false, b1v, (short)0, acc1, false, false);
    }

    // Epilogue: C/D layout -> lane holds N = lr, VGPR r holds M = r + hi*8.
    const float bias0 = bsum[nbase + lr];
    const float bias1 = bsum[nbase + 16 + lr];
    #pragma unroll
    for (int r = 0; r < 8; ++r) {
        int m = m0 + r + hi * 8;
        gates[(size_t)m * GATES + nbase + lr]      = acc0[r] + bias0;
        gates[(size_t)m * GATES + nbase + 16 + lr] = acc1[r] + bias1;
    }
}

// ---------------------------------------------------------------------------
// Skip-LSTM cell update + Wp reduction.  One 128-thread block per batch row.
// ---------------------------------------------------------------------------
__global__ __launch_bounds__(128) void cell_kernel(
        const float* __restrict__ gates,
        float* __restrict__ c,
        float* __restrict__ h,
        float* __restrict__ up,
        float* __restrict__ cum,
        const float* __restrict__ Wp,     // (1,128)
        const float* __restrict__ bp,     // (1,)
        _Float16* __restrict__ xh,        // write h_f16 into [:,192:320]
        float* __restrict__ out_h,        // (B,128) final hidden
        float* __restrict__ out_u,        // (B,) this step's u
        int write_h)
{
    const int b = blockIdx.x;
    const int t = threadIdx.x;
    __shared__ float red[HID];

    const float* grow = gates + (size_t)b * GATES;
    float gi = sigmoidf_(grow[t]);
    float gf = sigmoidf_(grow[HID + t]);
    float gg = tanhf(grow[2 * HID + t]);
    float go = sigmoidf_(grow[3 * HID + t]);

    float cp = c[(size_t)b * HID + t];
    float hp = h[(size_t)b * HID + t];
    float up_p = up[b];
    float cum_p = cum[b];

    float cumn = cum_p + fminf(up_p, 1.0f - cum_p);
    float u = rintf(cumn);              // forward value of straight-through round

    float nct = gf * cp + gi * gg;
    float nht = go * tanhf(nct);
    float nc = u * nct + (1.0f - u) * cp;
    float nh = u * nht + (1.0f - u) * hp;

    c[(size_t)b * HID + t] = nc;
    h[(size_t)b * HID + t] = nh;
    xh[(size_t)b * KTOT + INSZ + t] = (_Float16)nh;
    if (write_h) out_h[(size_t)b * HID + t] = nh;

    // dot(new_c, Wp)
    red[t] = nc * Wp[t];
    __syncthreads();
    for (int s = 64; s > 0; s >>= 1) {
        if (t < s) red[t] += red[t + s];
        __syncthreads();
    }
    if (t == 0) {
        float nupt = sigmoidf_(red[0] + bp[0]);
        up[b] = u * nupt + (1.0f - u) * up_p;
        cum[b] = (1.0f - u) * cumn;
        out_u[b] = u;
    }
}

// ---------------------------------------------------------------------------
extern "C" void kernel_launch(void* const* d_in, const int* in_sizes, int n_in,
                              void* d_out, int out_size, void* d_ws, size_t ws_size,
                              hipStream_t stream) {
    const float* image_pairs = (const float*)d_in[0];
    const float* Wih = (const float*)d_in[1];
    const float* Whh = (const float*)d_in[2];
    const float* bih = (const float*)d_in[3];
    const float* bhh = (const float*)d_in[4];
    const float* Wp  = (const float*)d_in[5];
    const float* bp  = (const float*)d_in[6];
    const float* Wg  = (const float*)d_in[7];
    const float* bg  = (const float*)d_in[8];

    const int B = in_sizes[0] / (2 * 3 * IMG * IMG);

    // Workspace partition (256B aligned)
    char* ws = (char*)d_ws;
    size_t cur = 0;
    auto take = [&](size_t bytes) -> char* {
        char* p = ws + cur;
        cur = (cur + bytes + 255) & ~(size_t)255;
        return p;
    };
    _Float16* Wc   = (_Float16*)take((size_t)GATES * KTOT * sizeof(_Float16));
    float*    bsum = (float*)   take((size_t)GATES * sizeof(float));
    _Float16* xh   = (_Float16*)take((size_t)B * KTOT * sizeof(_Float16));
    float*    gates= (float*)   take((size_t)B * GATES * sizeof(float));
    float*    cbuf = (float*)   take((size_t)B * HID * sizeof(float));
    float*    hbuf = (float*)   take((size_t)B * HID * sizeof(float));
    float*    upb  = (float*)   take((size_t)B * sizeof(float));
    float*    cumb = (float*)   take((size_t)B * sizeof(float));
    (void)ws_size;

    init_state_kernel<<<512, 256, 0, stream>>>(Wih, Whh, bih, bhh, Wc, bsum,
                                               xh, cbuf, hbuf, upb, cumb, B);

    float* out_h = (float*)d_out;
    float* out_u_base = (float*)d_out + (size_t)B * HID;

    const int NSTEPS = 16;   // NUM_GLIMPSES * 2 images
    for (int step = 0; step < NSTEPS; ++step) {
        int sel = step & 1;
        glimpse_kernel<<<B, 64, 0, stream>>>(image_pairs, sel, hbuf, Wg, bg, xh);
        dim3 ggrid(B / 16, GATES / 128);   // 4 waves/block, 2 N-tiles/wave
        gates_gemm_kernel<<<ggrid, 128, 0, stream>>>(xh, Wc, bsum, gates);
        cell_kernel<<<B, 128, 0, stream>>>(gates, cbuf, hbuf, upb, cumb,
                                           Wp, bp, xh, out_h,
                                           out_u_base + (size_t)step * B,
                                           step == NSTEPS - 1 ? 1 : 0);
    }
}
